// BinaryConv2d_56461640073300
// MI455X (gfx1250) — matrix operand order repaired and checked by reading, not compiled, not run on hardware
//
#include <hip/hip_runtime.h>
#include <hip/hip_bf16.h>

typedef _Float16 half_t;
typedef __attribute__((ext_vector_type(16))) _Float16 v16h;
typedef __attribute__((ext_vector_type(8)))  _Float16 v8h;
typedef __attribute__((ext_vector_type(8)))  float    v8f;

#define C_IN   128
#define C_OUT  256
#define HH     56
#define WW     56
#define NIMG   32
#define PIX_PER_IMG (HH*WW)          /* 3136   */
#define NPIX   (NIMG*PIX_PER_IMG)    /* 100352 */
#define W_CO_STRIDE (C_IN*9)         /* 1152   */

#define BM 128
#define BN 128
#define BK 32
#define LDT 40   /* LDS row stride in halfs: 32 + 8 pad (80B) -> conflict-free b128 */

#define KSTEPS 36  /* 9 taps * (128/BK) */

/* ------------------------------------------------------------------ */
/* Kernel 0: one-time weight binarize + pack: wpk[kpos][co][ci] (f16) */
/* ------------------------------------------------------------------ */
__global__ __launch_bounds__(256)
void pack_weights_kernel(const float* __restrict__ w, half_t* __restrict__ wpk)
{
    const int e    = blockIdx.x * 256 + threadIdx.x;   // < 9*256*128 = 294912
    const int kpos = e >> 15;                          // / (256*128)
    const int r    = e & 32767;
    const int co   = r >> 7;
    const int ci   = r & 127;
    const float v  = w[co * W_CO_STRIDE + ci * 9 + kpos];
    const float s  = (v > 0.0f) ? 1.0f : ((v < 0.0f) ? -1.0f : 0.0f);
    wpk[e] = (half_t)s;
}

/* ------------------------------------------------------------------ */
/* Kernel 1: main implicit-GEMM conv. A fragments straight from L2-   */
/* resident packed weights (no LDS); B staged via LDS double buffer.  */
/* ------------------------------------------------------------------ */
__global__ __launch_bounds__(256)
void bconv_wmma_packed_kernel(const float* __restrict__ x,
                              const half_t* __restrict__ wpk,
                              float* __restrict__ out)
{
    __shared__ __align__(16) half_t Bsm[2][BN * LDT];   // 20 KB total

    const int tid   = threadIdx.x;
    const int lane  = tid & 31;
    const int wave  = tid >> 5;         // 0..7
    const int wm    = wave & 3;         // M quadrant (32 rows each)
    const int wn    = wave >> 2;        // N half (64 cols each)
    const int lhalf = lane >> 4;        // 0: lanes 0-15, 1: lanes 16-31
    const int l16   = lane & 15;

    const int m_block = blockIdx.y * BM;
    const int n_block = blockIdx.x * BN;

    // staging-thread coordinates (fixed across k-steps)
    const int st_row  = tid & 127;      // B: pixel column
    const int st_zone = tid >> 7;       // which 16-wide ci chunk

    // pixel geometry for this thread's B column
    const int p     = n_block + st_row;
    const int n_img = p / PIX_PER_IMG;
    const int prem  = p - n_img * PIX_PER_IMG;
    const int oh    = prem / WW;
    const int ow    = prem - oh * WW;

    v8f acc[2][4] = {};

    auto load_B = [&](int buf, int step) {
        const int kpos  = step >> 2;
        const int cbase = (step & 3) * BK;
        const int kh    = kpos / 3;
        const int kw    = kpos - kh * 3;
        const int ih    = oh + kh - 1;
        const int iw    = ow + kw - 1;
        const bool valid = (ih >= 0) & (ih < HH) & (iw >= 0) & (iw < WW);
        const int ci0   = cbase + st_zone * 16;
        const float* xp = x + (((size_t)(n_img * C_IN + ci0) * HH + ih) * WW + iw);

        v8h h0{}, h1{};
#pragma unroll
        for (int j = 0; j < 8; ++j) {
            float v = valid ? xp[(size_t)j * PIX_PER_IMG] : 0.0f;
            h0[j] = (half_t)v;
        }
#pragma unroll
        for (int j = 0; j < 8; ++j) {
            float v = valid ? xp[(size_t)(j + 8) * PIX_PER_IMG] : 0.0f;
            h1[j] = (half_t)v;
        }
        v8h* dst = (v8h*)&Bsm[buf][st_row * LDT + st_zone * 16];
        dst[0] = h0;                       // ds_store_b128
        dst[1] = h1;                       // ds_store_b128
    };

    load_B(0, 0);
    __syncthreads();

    for (int s = 0; s < KSTEPS; ++s) {
        const int buf   = s & 1;
        const int kpos  = s >> 2;
        const int cbase = (s & 3) * BK;

        if (s + 1 < KSTEPS) load_B(buf ^ 1, s + 1);

        // A fragments: directly from packed global weights (L2/L0 resident).
        // Per-lane 16x32 f16 layout: lanes 0-15 K{0-7,16-23}; lanes 16-31 K{8-15,24-31}.
        v16h afr[2];
#pragma unroll
        for (int fm = 0; fm < 2; ++fm) {
            const int row = m_block + wm * 32 + fm * 16 + l16;
            const half_t* ap = wpk + (size_t)kpos * (C_OUT * C_IN)
                                   + (size_t)row * C_IN + cbase + lhalf * 8;
            union { v16h v; v8h h[2]; } u;
            u.h[0] = *(const v8h*)(ap);        // global_load_b128
            u.h[1] = *(const v8h*)(ap + 16);   // global_load_b128
            afr[fm] = u.v;
        }
        // B fragments from LDS: lanes 0-15 hold K{0-7,16-23} of column N, lanes 16-31 K{8-15,24-31}.
        v16h bfr[4];
#pragma unroll
        for (int fn = 0; fn < 4; ++fn) {
            const int col    = wn * 64 + fn * 16 + l16;
            const half_t* bp = &Bsm[buf][col * LDT + lhalf * 8];
            union { v16h v; v8h h[2]; } u;
            u.h[0] = *(const v8h*)(bp);
            u.h[1] = *(const v8h*)(bp + 16);
            bfr[fn] = u.v;
        }

#pragma unroll
        for (int fm = 0; fm < 2; ++fm)
#pragma unroll
            for (int fn = 0; fn < 4; ++fn)
                acc[fm][fn] = __builtin_amdgcn_wmma_f32_16x16x32_f16(
                    false, afr[fm], false, bfr[fn],
                    (short)0, acc[fm][fn], false, false);

        __syncthreads();
    }

    // epilogue: VGPR r -> lanes 0-15 M=r, lanes 16-31 M=r+8; N = lane%16
#pragma unroll
    for (int fn = 0; fn < 4; ++fn) {
        const int pp   = n_block + wn * 64 + fn * 16 + l16;
        const int oimg = pp / PIX_PER_IMG;
        const int orem = pp - oimg * PIX_PER_IMG;
        float* obase   = out + (size_t)oimg * C_OUT * PIX_PER_IMG + orem;
#pragma unroll
        for (int fm = 0; fm < 2; ++fm) {
            const int co0 = m_block + wm * 32 + fm * 16 + lhalf * 8;
#pragma unroll
            for (int r = 0; r < 8; ++r)
                obase[(size_t)(co0 + r) * PIX_PER_IMG] = acc[fm][fn][r];
        }
    }
}

/* ------------------------------------------------------------------ */
/* Fallback (no workspace): fully self-contained round-1 kernel       */
/* ------------------------------------------------------------------ */
__global__ __launch_bounds__(256)
void bconv_wmma_fallback_kernel(const float* __restrict__ x,
                                const float* __restrict__ w,
                                float* __restrict__ out)
{
    __shared__ __align__(16) half_t Asm[2][BM * LDT];
    __shared__ __align__(16) half_t Bsm[2][BN * LDT];

    const int tid   = threadIdx.x;
    const int lane  = tid & 31;
    const int wave  = tid >> 5;
    const int wm    = wave & 3;
    const int wn    = wave >> 2;
    const int lhalf = lane >> 4;
    const int l16   = lane & 15;

    const int m_block = blockIdx.y * BM;
    const int n_block = blockIdx.x * BN;

    const int st_row  = tid & 127;
    const int st_zone = tid >> 7;

    const int p     = n_block + st_row;
    const int n_img = p / PIX_PER_IMG;
    const int prem  = p - n_img * PIX_PER_IMG;
    const int oh    = prem / WW;
    const int ow    = prem - oh * WW;

    v8f acc[2][4] = {};

    auto load_tiles = [&](int buf, int step) {
        const int kpos  = step >> 2;
        const int cbase = (step & 3) * BK;
        const int kh    = kpos / 3;
        const int kw    = kpos - kh * 3;
        {
            const int ci0   = cbase + st_zone * 16;
            const float* wp = w + (m_block + st_row) * W_CO_STRIDE + ci0 * 9 + kpos;
            v8h h0{}, h1{};
#pragma unroll
            for (int j = 0; j < 8; ++j) {
                float v = wp[j * 9];
                h0[j] = (half_t)((v > 0.0f) ? 1.0f : ((v < 0.0f) ? -1.0f : 0.0f));
            }
#pragma unroll
            for (int j = 0; j < 8; ++j) {
                float v = wp[(j + 8) * 9];
                h1[j] = (half_t)((v > 0.0f) ? 1.0f : ((v < 0.0f) ? -1.0f : 0.0f));
            }
            v8h* dst = (v8h*)&Asm[buf][st_row * LDT + st_zone * 16];
            dst[0] = h0; dst[1] = h1;
        }
        {
            const int ih = oh + kh - 1;
            const int iw = ow + kw - 1;
            const bool valid = (ih >= 0) & (ih < HH) & (iw >= 0) & (iw < WW);
            const int ci0   = cbase + st_zone * 16;
            const float* xp = x + (((size_t)(n_img * C_IN + ci0) * HH + ih) * WW + iw);
            v8h h0{}, h1{};
#pragma unroll
            for (int j = 0; j < 8; ++j)
                h0[j] = (half_t)(valid ? xp[(size_t)j * PIX_PER_IMG] : 0.0f);
#pragma unroll
            for (int j = 0; j < 8; ++j)
                h1[j] = (half_t)(valid ? xp[(size_t)(j + 8) * PIX_PER_IMG] : 0.0f);
            v8h* dst = (v8h*)&Bsm[buf][st_row * LDT + st_zone * 16];
            dst[0] = h0; dst[1] = h1;
        }
    };

    load_tiles(0, 0);
    __syncthreads();

    for (int s = 0; s < KSTEPS; ++s) {
        const int buf = s & 1;
        if (s + 1 < KSTEPS) load_tiles(buf ^ 1, s + 1);

        v16h afr[2];
#pragma unroll
        for (int fm = 0; fm < 2; ++fm) {
            const int row    = wm * 32 + fm * 16 + l16;
            const half_t* ap = &Asm[buf][row * LDT + lhalf * 8];
            union { v16h v; v8h h[2]; } u;
            u.h[0] = *(const v8h*)(ap);
            u.h[1] = *(const v8h*)(ap + 16);
            afr[fm] = u.v;
        }
        v16h bfr[4];
#pragma unroll
        for (int fn = 0; fn < 4; ++fn) {
            const int col    = wn * 64 + fn * 16 + l16;
            const half_t* bp = &Bsm[buf][col * LDT + lhalf * 8];
            union { v16h v; v8h h[2]; } u;
            u.h[0] = *(const v8h*)(bp);
            u.h[1] = *(const v8h*)(bp + 16);
            bfr[fn] = u.v;
        }
#pragma unroll
        for (int fm = 0; fm < 2; ++fm)
#pragma unroll
            for (int fn = 0; fn < 4; ++fn)
                acc[fm][fn] = __builtin_amdgcn_wmma_f32_16x16x32_f16(
                    false, afr[fm], false, bfr[fn],
                    (short)0, acc[fm][fn], false, false);

        __syncthreads();
    }

#pragma unroll
    for (int fn = 0; fn < 4; ++fn) {
        const int pp   = n_block + wn * 64 + fn * 16 + l16;
        const int oimg = pp / PIX_PER_IMG;
        const int orem = pp - oimg * PIX_PER_IMG;
        float* obase   = out + (size_t)oimg * C_OUT * PIX_PER_IMG + orem;
#pragma unroll
        for (int fm = 0; fm < 2; ++fm) {
            const int co0 = m_block + wm * 32 + fm * 16 + lhalf * 8;
#pragma unroll
            for (int r = 0; r < 8; ++r)
                obase[(size_t)(co0 + r) * PIX_PER_IMG] = acc[fm][fn][r];
        }
    }
}

/* ------------------------------------------------------------------ */
extern "C" void kernel_launch(void* const* d_in, const int* in_sizes, int n_in,
                              void* d_out, int out_size, void* d_ws, size_t ws_size,
                              hipStream_t stream) {
    (void)in_sizes; (void)n_in; (void)out_size;
    const float* x = (const float*)d_in[0];
    const float* w = (const float*)d_in[1];
    float* out     = (float*)d_out;

    const size_t packed_bytes = (size_t)9 * C_OUT * C_IN * sizeof(half_t); // 576 KB

    dim3 grid(NPIX / BN, C_OUT / BM);   // (784, 2)
    dim3 block(256);

    if (d_ws != nullptr && ws_size >= packed_bytes) {
        half_t* wpk = (half_t*)d_ws;
        pack_weights_kernel<<<dim3((9 * C_OUT * C_IN) / 256), block, 0, stream>>>(w, wpk);
        bconv_wmma_packed_kernel<<<grid, block, 0, stream>>>(x, wpk, out);
    } else {
        bconv_wmma_fallback_kernel<<<grid, block, 0, stream>>>(x, w, out);
    }
}
